// GptOssExperts_43542378447341
// MI455X (gfx1250) — 8-bit matrix operand order repaired and checked by reading, compiled, not run on hardware
//
#include <hip/hip_runtime.h>
#include <stdint.h>

#define T_TOK 1024
#define NE    16
#define TK    4
#define HDIM  2048
#define I2    4096
#define IM    2048
#define CAP   512
#define FLIMIT 7.0f
#define FALPHA 1.702f

typedef int v8i __attribute__((ext_vector_type(8)));

__device__ __forceinline__ unsigned lds_off(const void* p) {
  return (unsigned)(unsigned long long)p;
}
__device__ __forceinline__ void async_copy_b128(unsigned lds, const void* g) {
  asm volatile("global_load_async_to_lds_b128 %0, %1, off"
               :: "v"(lds), "v"((unsigned long long)g) : "memory");
}
__device__ __forceinline__ void wait_async() {
  asm volatile("s_wait_asynccnt 0x0" ::: "memory");
}

__device__ __forceinline__ v8i loadA_frag(const signed char* base) {
  int4 lo = *(const int4*)(base);
  int4 hi = *(const int4*)(base + 32);
  v8i a;
  a[0] = lo.x; a[1] = lo.y; a[2] = lo.z; a[3] = lo.w;
  a[4] = hi.x; a[5] = hi.y; a[6] = hi.z; a[7] = hi.w;
  return a;
}
__device__ __forceinline__ v8i loadB_frag(const signed char* base) {
  int4 lo = *(const int4*)(base);
  int4 hi = *(const int4*)(base + 32);
  v8i b;
  b[0] = lo.x; b[1] = lo.y; b[2] = lo.z; b[3] = lo.w;
  b[4] = hi.x; b[5] = hi.y; b[6] = hi.z; b[7] = hi.w;
  return b;
}

__global__ __launch_bounds__(256) void k_cvtT(const float* __restrict__ src,
                                              signed char* __restrict__ dst,
                                              int Kd, int Nd) {
  __shared__ __align__(16) signed char Tls[64 * 80];
  const int e = blockIdx.z;
  const int kb = blockIdx.y * 64;
  const int n0 = blockIdx.x * 64;
  const int tid = threadIdx.x;
  const int kq = (tid >> 4) * 4;
  const int nl = (tid & 15) * 4;
  const float* s = src + (size_t)e * Kd * Nd + (size_t)(kb + kq) * Nd + (n0 + nl);
  int w_[4];
#pragma unroll
  for (int j = 0; j < 4; ++j) {
    float4 f = *(const float4*)(s + (size_t)j * Nd);
    int a = (int)rintf(f.x) & 255;
    int b = (int)rintf(f.y) & 255;
    int c = (int)rintf(f.z) & 255;
    int d = (int)rintf(f.w) & 255;
    w_[j] = a | (b << 8) | (c << 16) | (d << 24);
  }
#pragma unroll
  for (int i = 0; i < 4; ++i) {
    int sh = 8 * i;
    int v = ((w_[0] >> sh) & 255) | (((w_[1] >> sh) & 255) << 8) |
            (((w_[2] >> sh) & 255) << 16) | (((w_[3] >> sh) & 255) << 24);
    *(int*)(Tls + (nl + i) * 80 + kq) = v;
  }
  __syncthreads();
  const int row = tid >> 2;
  const int c4 = (tid & 3) * 16;
  *(int4*)(dst + (size_t)e * Kd * Nd + (size_t)(n0 + row) * Kd + kb + c4) =
      *(const int4*)(Tls + row * 80 + c4);
}

__global__ __launch_bounds__(256) void k_gate_quant(
    const float* __restrict__ hs, const float* __restrict__ logits,
    signed char* __restrict__ xq, float* __restrict__ xs,
    int* __restrict__ topk_e, float* __restrict__ topk_w) {
  const int t = blockIdx.x;
  const int tid = threadIdx.x;
  __shared__ float red[256];
  __shared__ float s_scale;

  const float* row = hs + (size_t)t * HDIM;
  float vals[8];
  float m = 0.f;
#pragma unroll
  for (int j = 0; j < 8; ++j) {
    vals[j] = row[tid * 8 + j];
    m = fmaxf(m, fabsf(vals[j]));
  }
  red[tid] = m;
  __syncthreads();
  for (int s = 128; s > 0; s >>= 1) {
    if (tid < s) red[tid] = fmaxf(red[tid], red[tid + s]);
    __syncthreads();
  }
  if (tid == 0) {
    s_scale = fmaxf(red[0] / 127.f, 1e-8f);
    float l[NE];
    float mx = -1e30f;
    for (int e = 0; e < NE; ++e) { l[e] = logits[t * NE + e]; mx = fmaxf(mx, l[e]); }
    float sum = 0.f;
    for (int e = 0; e < NE; ++e) { l[e] = __expf(l[e] - mx); sum += l[e]; }
    float inv = 1.f / sum;
    for (int e = 0; e < NE; ++e) l[e] *= inv;
    int idx[TK]; float wv[TK]; float wsum = 0.f;
    for (int k = 0; k < TK; ++k) {
      int best = 0; float bv = -1.f;
      for (int e = 0; e < NE; ++e) if (l[e] > bv) { bv = l[e]; best = e; }
      idx[k] = best; wv[k] = bv; l[best] = -1.f; wsum += bv;
    }
    float winv = 1.f / wsum;
    for (int k = 0; k < TK; ++k) {
      topk_e[t * TK + k] = idx[k];
      topk_w[t * TK + k] = wv[k] * winv;
    }
    xs[t] = s_scale;
  }
  __syncthreads();
  const float inv = 1.f / s_scale;
  int pk[2] = {0, 0};
#pragma unroll
  for (int j = 0; j < 8; ++j) {
    float q = fminf(fmaxf(rintf(vals[j] * inv), -127.f), 127.f);
    pk[j >> 2] |= ((int)q & 255) << (8 * (j & 3));
  }
  ((int2*)(xq + (size_t)t * HDIM))[tid] = make_int2(pk[0], pk[1]);
}

__global__ void k_route(const int* __restrict__ topk_e, int* __restrict__ slot) {
  const int e = threadIdx.x;
  if (e >= NE) return;
  int cnt = 0;
  for (int i = 0; i < T_TOK * TK; ++i) {
    if (topk_e[i] == e) {
      slot[i] = (cnt < CAP) ? (e * CAP + cnt) : -1;
      cnt++;
    }
  }
}

__global__ __launch_bounds__(128) void k_dispatch(
    const int* __restrict__ slot, const signed char* __restrict__ xq,
    const float* __restrict__ xs, signed char* __restrict__ xq_disp,
    float* __restrict__ s_disp) {
  const int i = blockIdx.x;
  const int s = slot[i];
  if (s < 0) return;
  const int src = i >> 2;
  ((int4*)(xq_disp + (size_t)s * HDIM))[threadIdx.x] =
      ((const int4*)(xq + (size_t)src * HDIM))[threadIdx.x];
  if (threadIdx.x == 0) s_disp[s] = xs[src];
}

__global__ __launch_bounds__(256) void k_gemm1(
    const signed char* __restrict__ xq_disp, const float* __restrict__ s_disp,
    const signed char* __restrict__ w8t, const float* __restrict__ wscale,
    const float* __restrict__ wbias, const float* __restrict__ smooth,
    float* __restrict__ act) {
  __shared__ __align__(16) signed char Als[2][64 * 80];
  __shared__ __align__(16) signed char Bgs[2][128 * 80];
  __shared__ __align__(16) signed char Bus[2][128 * 80];
  __shared__ float sxs[64];

  const int e = blockIdx.z;
  const int cblk = blockIdx.y;
  const int n0 = blockIdx.x * 128;

  const int tid = threadIdx.x;
  const int lane = tid & 31;
  const int w = tid >> 5;
  const int wm = w & 3;
  const int wn = w >> 2;
  const int m = lane & 15;
  const int half = lane >> 4;

  if (tid < 64) sxs[tid] = s_disp[e * CAP + cblk * 64 + tid];

  const v8i vz = {0, 0, 0, 0, 0, 0, 0, 0};
  v8i accg[4] = {vz, vz, vz, vz};
  v8i accu[4] = {vz, vz, vz, vz};

  const int arow = tid >> 2;
  const int acol = (tid & 3) * 16;
  const size_t abase = (size_t)(e * CAP + cblk * 64 + arow) * HDIM + acol;
  const int brow = tid >> 2;
  const int bcol = (tid & 3) * 16;
  const signed char* gB = w8t + ((size_t)e * I2 + n0) * HDIM;

  auto issue = [&](int buf, int kb) {
    async_copy_b128(lds_off(&Als[buf][arow * 80 + acol]), xq_disp + abase + kb);
    async_copy_b128(lds_off(&Bgs[buf][brow * 80 + bcol]),
                    gB + (size_t)brow * HDIM + kb + bcol);
    async_copy_b128(lds_off(&Bgs[buf][(brow + 64) * 80 + bcol]),
                    gB + (size_t)(brow + 64) * HDIM + kb + bcol);
    async_copy_b128(lds_off(&Bus[buf][brow * 80 + bcol]),
                    gB + (size_t)(IM + brow) * HDIM + kb + bcol);
    async_copy_b128(lds_off(&Bus[buf][(brow + 64) * 80 + bcol]),
                    gB + (size_t)(IM + brow + 64) * HDIM + kb + bcol);
  };

  issue(0, 0);
  for (int kb = 0, it = 0; kb < HDIM; kb += 64, ++it) {
    const int buf = it & 1;
    wait_async();
    __syncthreads();
    if (kb + 64 < HDIM) issue(buf ^ 1, kb + 64);

    const v8i a = loadA_frag(&Als[buf][(wm * 16 + m) * 80 + 16 * half]);
#pragma unroll
    for (int ti = 0; ti < 4; ++ti) {
      const v8i b = loadB_frag(&Bgs[buf][(wn * 64 + ti * 16 + m) * 80 + half * 16]);
      accg[ti] = __builtin_amdgcn_wmma_i32_16x16x64_iu8(true, a, true, b, accg[ti], false, false);
    }
#pragma unroll
    for (int ti = 0; ti < 4; ++ti) {
      const v8i b = loadB_frag(&Bus[buf][(wn * 64 + ti * 16 + m) * 80 + half * 16]);
      accu[ti] = __builtin_amdgcn_wmma_i32_16x16x64_iu8(true, a, true, b, accu[ti], false, false);
    }
    __syncthreads();
  }

#pragma unroll
  for (int ti = 0; ti < 4; ++ti) {
    const int ng = n0 + wn * 64 + ti * 16 + m;
    const float sg = wscale[e * I2 + ng];
    const float bgc = wbias[e * I2 + ng];
    const float su = wscale[e * I2 + ng + IM];
    const float buc = wbias[e * I2 + ng + IM];
    const float sm = smooth[e * IM + ng];
#pragma unroll
    for (int v = 0; v < 8; ++v) {
      const int rl = wm * 16 + v + 8 * half;
      const float sx = sxs[rl];
      float g = (float)accg[ti][v] * sx * sg + bgc;
      float u = (float)accu[ti][v] * sx * su + buc;
      g = fminf(g, FLIMIT);
      u = fminf(fmaxf(u, -FLIMIT), FLIMIT);
      float av = g * (1.0f / (1.0f + __expf(-FALPHA * g))) * (u + 1.0f) * sm;
      act[(size_t)(e * CAP + cblk * 64 + rl) * IM + ng] = av;
    }
  }
}

__global__ __launch_bounds__(256) void k_requant(const float* __restrict__ act,
                                                 signed char* __restrict__ hq,
                                                 float* __restrict__ hsc) {
  const int r = blockIdx.x;
  const int tid = threadIdx.x;
  const float* a = act + (size_t)r * IM;
  __shared__ float red[256];
  float vals[8];
  float m = 0.f;
#pragma unroll
  for (int j = 0; j < 8; ++j) {
    vals[j] = a[tid * 8 + j];
    m = fmaxf(m, fabsf(vals[j]));
  }
  red[tid] = m;
  __syncthreads();
  for (int s = 128; s > 0; s >>= 1) {
    if (tid < s) red[tid] = fmaxf(red[tid], red[tid + s]);
    __syncthreads();
  }
  const float sc = fmaxf(red[0] / 127.f, 1e-8f);
  const float inv = 1.f / sc;
  if (tid == 0) hsc[r] = sc;
  int pk[2] = {0, 0};
#pragma unroll
  for (int j = 0; j < 8; ++j) {
    float q = fminf(fmaxf(rintf(vals[j] * inv), -127.f), 127.f);
    pk[j >> 2] |= ((int)q & 255) << (8 * (j & 3));
  }
  ((int2*)(hq + (size_t)r * IM))[tid] = make_int2(pk[0], pk[1]);
}

__global__ __launch_bounds__(256) void k_gemm2(
    const signed char* __restrict__ hq, const float* __restrict__ hsc,
    const signed char* __restrict__ w8t, const float* __restrict__ wscale,
    const float* __restrict__ wbias, float* __restrict__ z) {
  __shared__ __align__(16) signed char Als[2][64 * 80];
  __shared__ __align__(16) signed char Bs[2][128 * 80];
  __shared__ float shs[64];

  const int e = blockIdx.z;
  const int cblk = blockIdx.y;
  const int n0 = blockIdx.x * 128;

  const int tid = threadIdx.x;
  const int lane = tid & 31;
  const int w = tid >> 5;
  const int wm = w & 3;
  const int wn = w >> 2;
  const int m = lane & 15;
  const int half = lane >> 4;

  if (tid < 64) shs[tid] = hsc[e * CAP + cblk * 64 + tid];

  const v8i vz = {0, 0, 0, 0, 0, 0, 0, 0};
  v8i acc[4] = {vz, vz, vz, vz};

  const int arow = tid >> 2;
  const int acol = (tid & 3) * 16;
  const size_t abase = (size_t)(e * CAP + cblk * 64 + arow) * IM + acol;
  const int brow = tid >> 2;
  const int bcol = (tid & 3) * 16;
  const signed char* gB = w8t + ((size_t)e * HDIM + n0) * IM;

  auto issue = [&](int buf, int kb) {
    async_copy_b128(lds_off(&Als[buf][arow * 80 + acol]), hq + abase + kb);
    async_copy_b128(lds_off(&Bs[buf][brow * 80 + bcol]),
                    gB + (size_t)brow * IM + kb + bcol);
    async_copy_b128(lds_off(&Bs[buf][(brow + 64) * 80 + bcol]),
                    gB + (size_t)(brow + 64) * IM + kb + bcol);
  };

  issue(0, 0);
  for (int kb = 0, it = 0; kb < IM; kb += 64, ++it) {
    const int buf = it & 1;
    wait_async();
    __syncthreads();
    if (kb + 64 < IM) issue(buf ^ 1, kb + 64);

    const v8i a = loadA_frag(&Als[buf][(wm * 16 + m) * 80 + 16 * half]);
#pragma unroll
    for (int ti = 0; ti < 4; ++ti) {
      const v8i b = loadB_frag(&Bs[buf][(wn * 64 + ti * 16 + m) * 80 + half * 16]);
      acc[ti] = __builtin_amdgcn_wmma_i32_16x16x64_iu8(true, a, true, b, acc[ti], false, false);
    }
    __syncthreads();
  }

#pragma unroll
  for (int ti = 0; ti < 4; ++ti) {
    const int ng = n0 + wn * 64 + ti * 16 + m;
    const float sw_ = wscale[e * HDIM + ng];
    const float bw_ = wbias[e * HDIM + ng];
#pragma unroll
    for (int v = 0; v < 8; ++v) {
      const int rl = wm * 16 + v + 8 * half;
      z[(size_t)(e * CAP + cblk * 64 + rl) * HDIM + ng] =
          (float)acc[ti][v] * shs[rl] * sw_ + bw_;
    }
  }
}

__global__ __launch_bounds__(256) void k_combine(const int* __restrict__ slot,
                                                 const float* __restrict__ topk_w,
                                                 const float* __restrict__ z,
                                                 float* __restrict__ out) {
  const int t = blockIdx.x;
  const int tid = threadIdx.x;
  __shared__ int ss[TK];
  __shared__ float sw[TK];
  if (tid < TK) {
    ss[tid] = slot[t * TK + tid];
    sw[tid] = topk_w[t * TK + tid];
  }
  __syncthreads();
  for (int c = tid; c < HDIM; c += 256) {
    float acc = 0.f;
#pragma unroll
    for (int k = 0; k < TK; ++k) {
      int s = ss[k];
      if (s >= 0) acc += sw[k] * z[(size_t)s * HDIM + c];
    }
    out[(size_t)t * HDIM + c] = acc;
  }
}

extern "C" void kernel_launch(void* const* d_in, const int* in_sizes, int n_in,
                              void* d_out, int out_size, void* d_ws, size_t ws_size,
                              hipStream_t stream) {
  const float* hs       = (const float*)d_in[0];
  const float* logits   = (const float*)d_in[1];
  const float* gup_f    = (const float*)d_in[2];
  const float* gu_scale = (const float*)d_in[3];
  const float* gu_bias  = (const float*)d_in[4];
  const float* gu_sm    = (const float*)d_in[5];
  const float* dn_f     = (const float*)d_in[6];
  const float* dn_scale = (const float*)d_in[7];
  const float* dn_bias  = (const float*)d_in[8];
  float* out = (float*)d_out;

  char* ws = (char*)d_ws;
  size_t off = 0;
  auto alloc = [&](size_t bytes) {
    size_t r = off;
    off = (off + bytes + 255) & ~(size_t)255;
    return r;
  };
  signed char* gu8t    = (signed char*)(ws + alloc((size_t)NE * HDIM * I2));
  signed char* dn8t    = (signed char*)(ws + alloc((size_t)NE * IM * HDIM));
  signed char* xq      = (signed char*)(ws + alloc((size_t)T_TOK * HDIM));
  float*       xs      = (float*)(ws + alloc((size_t)T_TOK * 4));
  int*         topk_e  = (int*)(ws + alloc((size_t)T_TOK * TK * 4));
  float*       topk_w  = (float*)(ws + alloc((size_t)T_TOK * TK * 4));
  int*         slot    = (int*)(ws + alloc((size_t)T_TOK * TK * 4));
  signed char* xq_disp = (signed char*)(ws + alloc((size_t)NE * CAP * HDIM));
  float*       s_disp  = (float*)(ws + alloc((size_t)NE * CAP * 4));
  float*       act     = (float*)(ws + alloc((size_t)NE * CAP * IM * 4));
  signed char* hq      = (signed char*)(ws + alloc((size_t)NE * CAP * IM));
  float*       hsc     = (float*)(ws + alloc((size_t)NE * CAP * 4));
  float*       z       = act;

  k_cvtT<<<dim3(I2 / 64, HDIM / 64, NE), 256, 0, stream>>>(gup_f, gu8t, HDIM, I2);
  k_cvtT<<<dim3(HDIM / 64, IM / 64, NE), 256, 0, stream>>>(dn_f, dn8t, IM, HDIM);
  k_gate_quant<<<T_TOK, 256, 0, stream>>>(hs, logits, xq, xs, topk_e, topk_w);
  k_route<<<1, 32, 0, stream>>>(topk_e, slot);
  k_dispatch<<<T_TOK * TK, 128, 0, stream>>>(slot, xq, xs, xq_disp, s_disp);
  k_gemm1<<<dim3(IM / 128, CAP / 64, NE), 256, 0, stream>>>(
      xq_disp, s_disp, gu8t, gu_scale, gu_bias, gu_sm, act);
  k_requant<<<NE * CAP, 256, 0, stream>>>(act, hq, hsc);
  k_gemm2<<<dim3(HDIM / 128, CAP / 64, NE), 256, 0, stream>>>(
      hq, hsc, dn8t, dn_scale, dn_bias, z);
  k_combine<<<T_TOK, 256, 0, stream>>>(slot, topk_w, z, out);
  (void)in_sizes; (void)n_in; (void)out_size; (void)ws_size;
}
